// LocalAttention2D_68298569941275
// MI455X (gfx1250) — matrix operand order repaired and checked by reading, not compiled, not run on hardware
//
#include <hip/hip_runtime.h>
#include <hip/hip_bf16.h>
#include <math.h>

// ---------------------------------------------------------------------------
// Fused windowed MHSA (Swin window attention) for MI455X / gfx1250.
//   x:[8,192,192,256] f32, w_qkv:[768,256], b_qkv:[768], w_proj:[256,256],
//   b_proj:[256]  ->  out:[8,192,192,256] f32
// One workgroup (256 threads = 8 wave32) per 8x8 window (4608 windows).
// All intermediates live in LDS; math in bf16 WMMA with f32 accumulate.
// V is stored TRANSPOSED in LDS so every WMMA operand (A and B) is loaded
// with aligned 16-byte ds_load_b128 pairs -- no strided scalar LDS reads.
// ---------------------------------------------------------------------------

typedef __bf16 bf16;
typedef __attribute__((ext_vector_type(16))) __bf16 v16bf;
typedef __attribute__((ext_vector_type(8)))  float  v8f;

#define NHEAD 8
#define CDIM  256
#define NTOK  64
#define IMG   192
#define NWS   24           // windows per side (192/8)
#define NWIN  (8*NWS*NWS)  // 4608

// LDS strides (bf16 elements), padded for bank-conflict-free row strides
#define XSTR  264          // 64 x 264   (X buffer, later attn-out buffer)
#define QKSTR 520          // 64 x 520   (Q,K buffer: channels 0..511)
#define TSTR  72           // 256 x 72   (V transposed: [channel][token])
#define PSTR  72           // 64 x 72 per wave (softmax P scratch)
#define LDS_ELEMS (NTOK*XSTR + NTOK*QKSTR + CDIM*TSTR + NHEAD*NTOK*PSTR)
#define LDS_BYTES (LDS_ELEMS * 2)

union FragU  { uint4 u[2]; v16bf v; };
union Pack8  { bf16 h[8];  uint4 u; };

// A fragment (16x32 bf16, M rows x K): lane = M row (lane&15),
// halves 0..7  -> K = k0 + koff + i        (koff = (lane>=16) ? 8 : 0)
// halves 8..15 -> K = k0 + 16 + koff + i
__device__ __forceinline__ v16bf ldA(const bf16* base, int stride, int m0,
                                     int k0, int lane) {
  const int row  = m0 + (lane & 15);
  const int koff = (lane >> 4) << 3;
  const bf16* p = base + row * stride + k0 + koff;
  FragU f;
  f.u[0] = *(const uint4*)(p);
  f.u[1] = *(const uint4*)(p + 16);
  return f.v;
}

// B fragment (32x16 bf16, K x N cols) from storage where element (k,n)
// lives at base[n*stride + k] (contiguous in k).
// lane = N col (lane&15), halves 0..7 -> K = k0 + koff + i, 8..15 -> K = k0 + 16 + koff + i (koff = (lane>=16) ? 8 : 0)
__device__ __forceinline__ v16bf ldBt(const bf16* base, int stride, int k0,
                                      int n0, int lane) {
  const int col = n0 + (lane & 15);
  const int kb  = k0 + ((lane >> 4) << 3);
  const bf16* p = base + col * stride + kb;
  FragU f;
  f.u[0] = *(const uint4*)(p);
  f.u[1] = *(const uint4*)(p + 16);
  return f.v;
}

__device__ __forceinline__ v8f wmma_bf16(v16bf a, v16bf b, v8f c) {
  return __builtin_amdgcn_wmma_f32_16x16x32_bf16(
      /*neg_a=*/false, a, /*neg_b=*/false, b,
      /*c_mod=*/(short)0, c, /*reuse_a=*/false, /*reuse_b=*/false);
}

__device__ __forceinline__ v8f zero8() {
  v8f z = {0.f, 0.f, 0.f, 0.f, 0.f, 0.f, 0.f, 0.f};
  return z;
}

// ---------------------------------------------------------------------------
// Pre-pass: convert fp32 weights to bf16 once into workspace.
// layout in ws: [0, 768*256) = w_qkv, [768*256, +256*256) = w_proj
// ---------------------------------------------------------------------------
__global__ void LocalAttn_cvt_weights(const float* __restrict__ wqkv,
                                      const float* __restrict__ wproj,
                                      bf16* __restrict__ out) {
  const int idx = blockIdx.x * blockDim.x + threadIdx.x;
  const int NQ = 768 * 256;
  const int NP = 256 * 256;
  if (idx < NQ)            out[idx] = (bf16)wqkv[idx];
  else if (idx < NQ + NP)  out[idx] = (bf16)wproj[idx - NQ];
}

// ---------------------------------------------------------------------------
// Main fused kernel: one block per window.
// ---------------------------------------------------------------------------
__global__ void __launch_bounds__(256)
LocalAttention2D_68298569941275_kernel(const float* __restrict__ x,
                                       const bf16* __restrict__ wqkv_bf,
                                       const bf16* __restrict__ wproj_bf,
                                       const float* __restrict__ b_qkv,
                                       const float* __restrict__ b_proj,
                                       float* __restrict__ out) {
  extern __shared__ __align__(16) char smem_raw[];
  bf16* xbuf = (bf16*)smem_raw;            // 64 x XSTR  (X, later attn-out)
  bf16* qkb  = xbuf + NTOK * XSTR;         // 64 x QKSTR (Q,K, bias folded)
  bf16* vt   = qkb  + NTOK * QKSTR;        // 256 x TSTR (V transposed)
  bf16* pbuf = vt   + CDIM * TSTR;         // 8 waves x (64 x PSTR)

  const int win  = blockIdx.x;
  const int bimg = win / (NWS * NWS);
  const int rem  = win % (NWS * NWS);
  const int wh   = rem / NWS;
  const int ww   = rem % NWS;

  const int tid  = threadIdx.x;
  const int wid  = tid >> 5;
  const int lane = tid & 31;
  const int rowoff = (lane >> 4) << 3;     // D-layout: +8 rows in upper half
  const int cb     = lane & 15;            // D-layout: N column

  // ---- Stage A: gather window tokens, f32 -> bf16 into LDS ---------------
  for (int i = tid; i < NTOK * 64; i += 256) {   // 64 float4 per token
    const int token = i >> 6;
    const int c4    = (i & 63) << 2;
    const int gr = wh * 8 + (token >> 3);
    const int gc = ww * 8 + (token & 7);
    const float4 v = *(const float4*)(x + (((size_t)bimg * IMG + gr) * IMG + gc) * CDIM + c4);
    bf16* dst = xbuf + token * XSTR + c4;
    dst[0] = (bf16)v.x; dst[1] = (bf16)v.y;
    dst[2] = (bf16)v.z; dst[3] = (bf16)v.w;
  }
  __syncthreads();

  // ---- Stage B: QKV = X @ Wqkv^T + b  (64x256 @ 256x768) -----------------
  // tile grid: 4 (M) x 48 (N); 24 tiles per wave; K = 8 steps of 32.
  // Q,K tiles (nt < 32) -> qkb row-major; V tiles (nt >= 32) -> vt transposed
  // (transposed store packs 8 consecutive rows into one ds_store_b128).
  for (int t = 0; t < 24; ++t) {
    const int tile = wid * 24 + t;
    const int mt = tile & 3;
    const int nt = tile >> 2;
    v8f acc = zero8();
#pragma unroll
    for (int kk = 0; kk < 8; ++kk) {
      const v16bf a = ldA(xbuf, XSTR, mt * 16, kk * 32, lane);
      const v16bf b = ldBt(wqkv_bf, CDIM, kk * 32, nt * 16, lane);
      acc = wmma_bf16(a, b, acc);
    }
    const int n     = nt * 16 + cb;
    const float bia = b_qkv[n];
    if (nt < 32) {
      // Q or K: scatter 8 bf16 rows (row-major token x channel layout)
#pragma unroll
      for (int r = 0; r < 8; ++r)
        qkb[(mt * 16 + r + rowoff) * QKSTR + n] = (bf16)(acc[r] + bia);
    } else {
      // V: packed store into transposed layout vt[channel][token]
      const int c     = n - 2 * CDIM;      // channel 0..255
      const int mbase = mt * 16 + rowoff;  // first of 8 consecutive tokens
      Pack8 pk;
#pragma unroll
      for (int r = 0; r < 8; ++r) pk.h[r] = (bf16)(acc[r] + bia);
      *(uint4*)(vt + c * TSTR + mbase) = pk.u;
    }
  }
  __syncthreads();

  // ---- Stage C: per-head attention; wave `wid` owns head `wid` -----------
  {
    const int h = wid;
    const bf16* Qb  = qkb + h * 32;          // Q cols  h*32 .. +31
    const bf16* Kb  = qkb + CDIM + h * 32;   // K cols  256 + h*32
    const bf16* VTh = vt + (h * 32) * TSTR;  // V^T rows = head channels
    bf16* Pw = pbuf + wid * NTOK * PSTR;

    // S = (Q K^T) * d^-0.5 : 4x4 tiles, K = d = 32 -> one WMMA per tile
    v8f S[4][4];
#pragma unroll
    for (int mt = 0; mt < 4; ++mt) {
      const v16bf qa = ldA(Qb, QKSTR, mt * 16, 0, lane);
#pragma unroll
      for (int nt = 0; nt < 4; ++nt) {
        const v16bf kb = ldBt(Kb, QKSTR, 0, nt * 16, lane);
        S[mt][nt] = wmma_bf16(qa, kb, zero8());
      }
    }

    // softmax over rows (row = one VGPR across a 16-lane half)
    const float scale = 0.17677669529663687f;   // 32^-0.5
#pragma unroll
    for (int mt = 0; mt < 4; ++mt) {
#pragma unroll
      for (int r = 0; r < 8; ++r) {
        float mx = -3.0e38f;
#pragma unroll
        for (int nt = 0; nt < 4; ++nt) {
          S[mt][nt][r] *= scale;
          mx = fmaxf(mx, S[mt][nt][r]);
        }
#pragma unroll
        for (int off = 1; off < 16; off <<= 1)
          mx = fmaxf(mx, __shfl_xor(mx, off, 32));
        float sum = 0.f;
#pragma unroll
        for (int nt = 0; nt < 4; ++nt) {
          const float e = __expf(S[mt][nt][r] - mx);
          S[mt][nt][r] = e;
          sum += e;
        }
#pragma unroll
        for (int off = 1; off < 16; off <<= 1)
          sum += __shfl_xor(sum, off, 32);
        const float inv = 1.0f / sum;
#pragma unroll
        for (int nt = 0; nt < 4; ++nt) S[mt][nt][r] *= inv;
      }
      // spill normalized P (bf16) into per-wave LDS scratch in A-major order
#pragma unroll
      for (int nt = 0; nt < 4; ++nt)
#pragma unroll
        for (int r = 0; r < 8; ++r)
          Pw[(mt * 16 + r + rowoff) * PSTR + nt * 16 + cb] = (bf16)S[mt][nt][r];
    }

    // O = P @ V : M=64, N=32 (2 tiles), K=64 (2 steps of 32).
    // B fragment (k=token, n=dim) read from vt[n*TSTR + k]: contiguous in k.
#pragma unroll
    for (int nt = 0; nt < 2; ++nt) {
#pragma unroll
      for (int mt = 0; mt < 4; ++mt) {
        v8f acc = zero8();
#pragma unroll
        for (int kk = 0; kk < 2; ++kk) {
          const v16bf a = ldA(Pw, PSTR, mt * 16, kk * 32, lane);
          const v16bf b = ldBt(VTh, TSTR, kk * 32, nt * 16, lane);
          acc = wmma_bf16(a, b, acc);
        }
        const int col = h * 32 + nt * 16 + cb;   // head-local dim -> channel
#pragma unroll
        for (int r = 0; r < 8; ++r)
          xbuf[(mt * 16 + r + rowoff) * XSTR + col] = (bf16)acc[r];
      }
    }
  }
  __syncthreads();

  // ---- Stage D: out = O @ Wproj^T + b, scatter to unpartitioned layout ---
  // N = 256 -> 16 tiles, 2 per wave; M = 4 tiles; K = 8 steps of 32
  for (int nn = 0; nn < 2; ++nn) {
    const int nt = wid * 2 + nn;
    for (int mt = 0; mt < 4; ++mt) {
      v8f acc = zero8();
#pragma unroll
      for (int kk = 0; kk < 8; ++kk) {
        const v16bf a = ldA(xbuf, XSTR, mt * 16, kk * 32, lane);
        const v16bf b = ldBt(wproj_bf, CDIM, kk * 32, nt * 16, lane);
        acc = wmma_bf16(a, b, acc);
      }
      const int n     = nt * 16 + cb;
      const float bia = b_proj[n];
#pragma unroll
      for (int r = 0; r < 8; ++r) {
        const int m  = mt * 16 + r + rowoff;
        const int gr = wh * 8 + (m >> 3);
        const int gc = ww * 8 + (m & 7);
        out[(((size_t)bimg * IMG + gr) * IMG + gc) * CDIM + n] = acc[r] + bia;
      }
    }
  }
}

// ---------------------------------------------------------------------------
extern "C" void kernel_launch(void* const* d_in, const int* in_sizes, int n_in,
                              void* d_out, int out_size, void* d_ws,
                              size_t ws_size, hipStream_t stream) {
  (void)in_sizes; (void)n_in; (void)out_size; (void)ws_size;

  const float* x      = (const float*)d_in[0];
  const float* w_qkv  = (const float*)d_in[1];
  const float* b_qkv  = (const float*)d_in[2];
  const float* w_proj = (const float*)d_in[3];
  const float* b_proj = (const float*)d_in[4];
  float*       out    = (float*)d_out;

  bf16* wbf      = (bf16*)d_ws;                 // 768*256 + 256*256 bf16
  bf16* wproj_bf = wbf + 768 * 256;

  // cheap weight conversion (re-run every call: deterministic)
  const int NW = 768 * 256 + 256 * 256;         // 262144
  LocalAttn_cvt_weights<<<(NW + 255) / 256, 256, 0, stream>>>(w_qkv, w_proj, wbf);

  // allow >64KB dynamic LDS (CDNA5 WGP has 320KB)
  hipFuncSetAttribute((const void*)LocalAttention2D_68298569941275_kernel,
                      hipFuncAttributeMaxDynamicSharedMemorySize, LDS_BYTES);

  LocalAttention2D_68298569941275_kernel<<<NWIN, 256, LDS_BYTES, stream>>>(
      x, wbf, wproj_bf, b_qkv, b_proj, out);
}
